// Attention_67637144978498
// MI455X (gfx1250) — matrix operand order repaired and checked by reading, not compiled, not run on hardware
//
#include <hip/hip_runtime.h>
#include <hip/hip_bf16.h>

// ---------- types ----------
typedef __attribute__((ext_vector_type(16))) __bf16 v16bf;
typedef __attribute__((ext_vector_type(8)))  float  v8f;

union Frag {
    v16bf v;
    uint4 q[2];
};

__device__ __forceinline__ unsigned short f2bf(float f) {
    unsigned u = __float_as_uint(f);
    u += 0x7FFFu + ((u >> 16) & 1u);   // round-to-nearest-even
    return (unsigned short)(u >> 16);
}

__device__ __forceinline__ void async_b128(unsigned lds_addr, const void* gaddr) {
    asm volatile("global_load_async_to_lds_b128 %0, %1, off"
                 :: "v"(lds_addr), "v"((unsigned long long)(uintptr_t)gaddr)
                 : "memory");
}

__device__ __forceinline__ void wait_async0() {
#if __has_builtin(__builtin_amdgcn_s_wait_asynccnt)
    __builtin_amdgcn_s_wait_asynccnt(0);
#else
    asm volatile("s_wait_asynccnt 0" ::: "memory");
#endif
}

// ---------- constants ----------
#define BB   2
#define LL   2048
#define DD   1024
#define NH   16
#define HH   64
#define MM   (BB * LL)      // 4096 rows

// ================= fp32 -> bf16 bulk convert =================
__global__ __launch_bounds__(256) void f32_to_bf16_kernel(
    const float* __restrict__ src, unsigned short* __restrict__ dst) {
    int i = (blockIdx.x * 256 + threadIdx.x) * 4;
    const float4 v = *reinterpret_cast<const float4*>(&src[i]);
    ushort4 h;
    h.x = f2bf(v.x); h.y = f2bf(v.y); h.z = f2bf(v.z); h.w = f2bf(v.w);
    *reinterpret_cast<ushort4*>(&dst[i]) = h;
}

// ====== GEMM: Y = A * W^T + bias (bf16 A/W; 128x128 tile, 8 waves; async dbl-buffered LDS) ======
#define TILE_M 128
#define TILE_N 128
#define TILE_K 32
#define NKT    (DD / TILE_K)   // 32
#define LDA 40                 // padded LDS row stride (elements)
#define LDB 40

__global__ __launch_bounds__(256) void gemm_bias_kernel(
    const unsigned short* __restrict__ A, const unsigned short* __restrict__ Bw,
    const float* __restrict__ bias, float* __restrict__ Y) {
    __shared__ alignas(16) unsigned short As[2][TILE_M * LDA];
    __shared__ alignas(16) unsigned short Bs[2][TILE_N * LDB];

    const int tid  = threadIdx.x;
    const int lane = tid & 31;
    const int wave = tid >> 5;          // 0..7
    const int row0 = blockIdx.y * TILE_M;
    const int col0 = blockIdx.x * TILE_N;
    const int wm = wave & 1;            // 2 row groups of 64
    const int wn = wave >> 1;           // 4 col groups of 32

    const int m_lane = lane & 15;
    const int kgrp   = lane >> 4;       // half-wave
    const int akb    = kgrp * 8;        // A-frag K chunk base
    const int bkb    = kgrp * 8;        // B-frag K chunk base

    // staging: 128x32 tile = 512 chunks of 8 bf16; 256 threads x 2 chunks
    const int st_r = tid >> 2;          // 0..63
    const int st_c = (tid & 3) << 3;    // 0,8,16,24

    auto issue_tile = [&](int buf, int k0) {
        for (int i = 0; i < 2; ++i) {
            int r = st_r + i * 64;
            async_b128((unsigned)(uintptr_t)&As[buf][r * LDA + st_c],
                       &A[(size_t)(row0 + r) * DD + k0 + st_c]);
        }
        for (int i = 0; i < 2; ++i) {
            int r = st_r + i * 64;
            async_b128((unsigned)(uintptr_t)&Bs[buf][r * LDB + st_c],
                       &Bw[(size_t)(col0 + r) * DD + k0 + st_c]);
        }
    };

    v8f acc[4][2] = {};

    issue_tile(0, 0);
    for (int kt = 0; kt < NKT; ++kt) {
        const int cur = kt & 1;
        wait_async0();
        __syncthreads();                       // tile `cur` visible to all waves
        if (kt + 1 < NKT) issue_tile(cur ^ 1, (kt + 1) * TILE_K);

        Frag a[4], b[2];
        for (int mt = 0; mt < 4; ++mt) {
            const unsigned short* src = &As[cur][(wm * 64 + mt * 16 + m_lane) * LDA];
            a[mt].q[0] = *reinterpret_cast<const uint4*>(&src[akb]);
            a[mt].q[1] = *reinterpret_cast<const uint4*>(&src[akb + 16]);
        }
        for (int nt = 0; nt < 2; ++nt) {
            const unsigned short* src = &Bs[cur][(wn * 32 + nt * 16 + m_lane) * LDB];
            b[nt].q[0] = *reinterpret_cast<const uint4*>(&src[bkb]);
            b[nt].q[1] = *reinterpret_cast<const uint4*>(&src[bkb + 16]);
        }
        for (int mt = 0; mt < 4; ++mt)
            for (int nt = 0; nt < 2; ++nt)
                acc[mt][nt] = __builtin_amdgcn_wmma_f32_16x16x32_bf16(
                    false, a[mt].v, false, b[nt].v, (short)0, acc[mt][nt], false, false);
    }

    for (int mt = 0; mt < 4; ++mt) {
        for (int nt = 0; nt < 2; ++nt) {
            int col = col0 + wn * 32 + nt * 16 + m_lane;
            float bv = bias[col];
            for (int r = 0; r < 8; ++r) {
                int row = row0 + wm * 64 + mt * 16 + kgrp * 8 + r;
                Y[(size_t)row * DD + col] = acc[mt][nt][r] + bv;
            }
        }
    }
}

// ================= RoPE + repack: q,k -> [B,N,L,H] bf16 ; v -> [B,N,H,L] bf16 =================
__global__ __launch_bounds__(256) void rope_pack_kernel(
    const float* __restrict__ q, const float* __restrict__ k, const float* __restrict__ v,
    const float* __restrict__ scale_p,
    unsigned short* __restrict__ qb, unsigned short* __restrict__ kb,
    unsigned short* __restrict__ vt) {
    int idx = blockIdx.x * blockDim.x + threadIdx.x;   // < B*L*N*(H/2) = 2^21
    int h2 = idx & 31;            // H/2 = 32
    int n  = (idx >> 5) & 15;
    int l  = (idx >> 9) & 2047;
    int b  = idx >> 20;
    float scale = scale_p[0];

    size_t src = ((size_t)b * LL + l) * DD + n * HH + 2 * h2;
    // freqs = THETA^(-(2*h2)/H); ln(10000) = 9.210340371976184
    float freq = __expf(-((float)(2 * h2) * (1.0f / (float)HH)) * 9.2103403719761836f);
    float ang = (float)l * freq;
    float s, c;
    __sincosf(ang, &s, &c);

    float qe = q[src], qo = q[src + 1];
    float ke = k[src], ko = k[src + 1];
    float oqe = (c * qe - s * qo) * scale;
    float oqo = (s * qe + c * qo) * scale;
    float oke = c * ke - s * ko;
    float oko = s * ke + c * ko;

    int bn = b * NH + n;
    size_t dq = ((size_t)bn * LL + l) * HH + 2 * h2;
    qb[dq] = f2bf(oqe); qb[dq + 1] = f2bf(oqo);
    kb[dq] = f2bf(oke); kb[dq + 1] = f2bf(oko);

    float ve = v[src], vo = v[src + 1];
    size_t dv = ((size_t)bn * HH + 2 * h2) * LL + l;
    vt[dv]      = f2bf(ve);
    vt[dv + LL] = f2bf(vo);
}

// ================= Flash attention (causal), one head / 64 query rows per block =================
#define ATT_LDK 72   // 64 + 8 pad
#define ATT_LDV 40   // 32 + 8 pad
#define ATT_LDP 40

__global__ __launch_bounds__(128) void flash_attn_kernel(
    const unsigned short* __restrict__ qb, const unsigned short* __restrict__ kb,
    const unsigned short* __restrict__ vt, unsigned short* __restrict__ outb) {
    __shared__ alignas(16) unsigned short Ks[32 * ATT_LDK];        // keys x feats
    __shared__ alignas(16) unsigned short Vts[64 * ATT_LDV];       // feats x keys
    __shared__ alignas(16) unsigned short Ps[4 * 16 * ATT_LDP];    // per-wave P scratch

    const int tid  = threadIdx.x;
    const int lane = tid & 31;
    const int wave = tid >> 5;
    const int bn   = blockIdx.y;                       // b*N + n
    const int q0   = blockIdx.x * 64 + wave * 16;
    const int m_lane = lane & 15;
    const int kgrp   = lane >> 4;
    const int akb    = kgrp * 8;
    const int bkb    = kgrp * 8;

    // Q fragments: 16 rows x 64 feats -> two K=32 A-fragments
    Frag aq[2];
    {
        const unsigned short* qrow = &qb[((size_t)bn * LL + q0 + m_lane) * HH];
        aq[0].q[0] = *reinterpret_cast<const uint4*>(&qrow[akb]);
        aq[0].q[1] = *reinterpret_cast<const uint4*>(&qrow[akb + 16]);
        aq[1].q[0] = *reinterpret_cast<const uint4*>(&qrow[32 + akb]);
        aq[1].q[1] = *reinterpret_cast<const uint4*>(&qrow[32 + akb + 16]);
    }

    v8f o_acc[4] = {};
    float m_st[8], l_st[8];
    for (int r = 0; r < 8; ++r) { m_st[r] = -1e30f; l_st[r] = 0.0f; }

    const int q_end = q0 + 15;
    const int jmax  = blockIdx.x * 64 + 64;
    unsigned short* Pw = &Ps[wave * 16 * ATT_LDP];

    for (int j0 = 0; j0 < jmax; j0 += 32) {
        // async stage: K tile (32 keys x 64 feats), V^T tile (64 feats x 32 keys)
        for (int i = 0; i < 2; ++i) {
            int cid = tid + i * 128;          // 256 chunks of 8 bf16
            int r   = cid >> 3;
            int cof = (cid & 7) << 3;
            async_b128((unsigned)(uintptr_t)&Ks[r * ATT_LDK + cof],
                       &kb[((size_t)bn * LL + j0 + r) * HH + cof]);
        }
        for (int i = 0; i < 2; ++i) {
            int cid = tid + i * 128;
            int f   = cid >> 2;
            int cof = (cid & 3) << 3;
            async_b128((unsigned)(uintptr_t)&Vts[f * ATT_LDV + cof],
                       &vt[((size_t)bn * HH + f) * LL + j0 + cof]);
        }
        wait_async0();
        __syncthreads();

        if (j0 <= q_end) {
            // S = Q K^T : 16 x 32, two 16x16 tiles, K=64 in two WMMAs each
            v8f s[2] = {};
            for (int nt = 0; nt < 2; ++nt) {
                for (int kw = 0; kw < 2; ++kw) {
                    Frag bkf;
                    const unsigned short* src = &Ks[(nt * 16 + m_lane) * ATT_LDK + kw * 32 + bkb];
                    bkf.q[0] = *reinterpret_cast<const uint4*>(&src[0]);
                    bkf.q[1] = *reinterpret_cast<const uint4*>(&src[16]);
                    s[nt] = __builtin_amdgcn_wmma_f32_16x16x32_bf16(
                        false, aq[kw].v, false, bkf.v, (short)0, s[nt], false, false);
                }
            }
            // causal mask + online softmax (row = half-wave of 16 lanes)
            float p0v[8], p1v[8];
            for (int r = 0; r < 8; ++r) {
                int rowq = q0 + kgrp * 8 + r;
                float s0 = s[0][r];
                float s1 = s[1][r];
                if (j0 + m_lane > rowq)      s0 = -1e30f;
                if (j0 + 16 + m_lane > rowq) s1 = -1e30f;
                float rmax = fmaxf(s0, s1);
                rmax = fmaxf(rmax, __shfl_xor(rmax, 1, 32));
                rmax = fmaxf(rmax, __shfl_xor(rmax, 2, 32));
                rmax = fmaxf(rmax, __shfl_xor(rmax, 4, 32));
                rmax = fmaxf(rmax, __shfl_xor(rmax, 8, 32));
                float m_new = fmaxf(m_st[r], rmax);
                float corr  = __expf(m_st[r] - m_new);
                float p0 = __expf(s0 - m_new);
                float p1 = __expf(s1 - m_new);
                float rsum = p0 + p1;
                rsum += __shfl_xor(rsum, 1, 32);
                rsum += __shfl_xor(rsum, 2, 32);
                rsum += __shfl_xor(rsum, 4, 32);
                rsum += __shfl_xor(rsum, 8, 32);
                l_st[r] = l_st[r] * corr + rsum;
                m_st[r] = m_new;
                for (int ft = 0; ft < 4; ++ft) o_acc[ft][r] *= corr;
                p0v[r] = p0; p1v[r] = p1;
            }
            // P: C-layout -> LDS -> A-fragment
            for (int r = 0; r < 8; ++r) {
                int rg = kgrp * 8 + r;
                Pw[rg * ATT_LDP + m_lane]      = f2bf(p0v[r]);
                Pw[rg * ATT_LDP + 16 + m_lane] = f2bf(p1v[r]);
            }
            asm volatile("s_wait_dscnt 0" ::: "memory");   // wave-local LDS RAW (cross-lane)
            Frag ap;
            {
                const unsigned short* src = &Pw[m_lane * ATT_LDP];
                ap.q[0] = *reinterpret_cast<const uint4*>(&src[akb]);
                ap.q[1] = *reinterpret_cast<const uint4*>(&src[akb + 16]);
            }
            // O += P V : 16x64 in four 16x16 tiles (K=32)
            for (int ft = 0; ft < 4; ++ft) {
                Frag bv;
                const unsigned short* src = &Vts[(ft * 16 + m_lane) * ATT_LDV + bkb];
                bv.q[0] = *reinterpret_cast<const uint4*>(&src[0]);
                bv.q[1] = *reinterpret_cast<const uint4*>(&src[16]);
                o_acc[ft] = __builtin_amdgcn_wmma_f32_16x16x32_bf16(
                    false, ap.v, false, bv.v, (short)0, o_acc[ft], false, false);
            }
        }
        __syncthreads();
    }

    // normalize + write O (bf16) back into [B, L, D] at head slot
    float inv_l[8];
    for (int r = 0; r < 8; ++r) inv_l[r] = 1.0f / l_st[r];
    int b = bn >> 4, n = bn & 15;
    for (int ft = 0; ft < 4; ++ft) {
        int col = n * HH + ft * 16 + m_lane;
        for (int r = 0; r < 8; ++r) {
            int row = q0 + kgrp * 8 + r;
            outb[((size_t)b * LL + row) * DD + col] = f2bf(o_acc[ft][r] * inv_l[r]);
        }
    }
}

// ================= host launch =================
extern "C" void kernel_launch(void* const* d_in, const int* in_sizes, int n_in,
                              void* d_out, int out_size, void* d_ws, size_t ws_size,
                              hipStream_t stream) {
    (void)in_sizes; (void)n_in; (void)out_size; (void)ws_size;
    const float* x     = (const float*)d_in[0];
    const float* Wq    = (const float*)d_in[1];
    const float* bq    = (const float*)d_in[2];
    const float* Wk    = (const float*)d_in[3];
    const float* bk_   = (const float*)d_in[4];
    const float* Wv    = (const float*)d_in[5];
    const float* bv    = (const float*)d_in[6];
    const float* Wo    = (const float*)d_in[7];
    const float* bo    = (const float*)d_in[8];
    const float* scale = (const float*)d_in[9];
    float* out = (float*)d_out;

    char* ws = (char*)d_ws;
    const size_t E  = (size_t)BB * LL * DD;        // 4 Mi elements (x-shaped)
    const size_t WE = (size_t)DD * DD;             // 1 Mi elements (weight-shaped)
    float*          q     = (float*)(ws);                        // 16 MiB
    float*          k     = (float*)(ws + E * 4);                // 16 MiB
    float*          v     = (float*)(ws + E * 8);                // 16 MiB
    unsigned short* xb    = (unsigned short*)(ws + E * 12);      //  8 MiB
    unsigned short* wqb   = (unsigned short*)(ws + E * 14);      //  2 MiB
    unsigned short* wkb   = (unsigned short*)(ws + E * 14 + WE * 2);
    unsigned short* wvb   = (unsigned short*)(ws + E * 14 + WE * 4);
    unsigned short* wob   = (unsigned short*)(ws + E * 14 + WE * 6);
    unsigned short* qbuf  = (unsigned short*)(ws + E * 16);      //  8 MiB
    unsigned short* kbuf  = (unsigned short*)(ws + E * 18);      //  8 MiB
    unsigned short* vt    = (unsigned short*)(ws + E * 20);      //  8 MiB
    unsigned short* attnb = (unsigned short*)(ws + E * 22);      //  8 MiB

    // bf16 conversions (once)
    f32_to_bf16_kernel<<<(unsigned)(E / 1024), 256, 0, stream>>>(x, xb);
    f32_to_bf16_kernel<<<(unsigned)(WE / 1024), 256, 0, stream>>>(Wq, wqb);
    f32_to_bf16_kernel<<<(unsigned)(WE / 1024), 256, 0, stream>>>(Wk, wkb);
    f32_to_bf16_kernel<<<(unsigned)(WE / 1024), 256, 0, stream>>>(Wv, wvb);
    f32_to_bf16_kernel<<<(unsigned)(WE / 1024), 256, 0, stream>>>(Wo, wob);

    dim3 gg(DD / TILE_N, MM / TILE_M);   // (8, 32)
    gemm_bias_kernel<<<gg, 256, 0, stream>>>(xb, wqb, bq, q);
    gemm_bias_kernel<<<gg, 256, 0, stream>>>(xb, wkb, bk_, k);
    gemm_bias_kernel<<<gg, 256, 0, stream>>>(xb, wvb, bv, v);

    rope_pack_kernel<<<(unsigned)((E / 2) / 256), 256, 0, stream>>>(q, k, v, scale, qbuf, kbuf, vt);

    flash_attn_kernel<<<dim3(LL / 64, BB * NH), 128, 0, stream>>>(qbuf, kbuf, vt, attnb);

    gemm_bias_kernel<<<gg, 256, 0, stream>>>(attnb, wob, bo, out);
}
